// PraxisRecurrent_29480655519866
// MI455X (gfx1250) — compile-verified, matrix operand order checked by reading, not run on hardware
//
#include <hip/hip_runtime.h>
#include <cstdint>
#include <cstddef>

// Shapes (hardcoded from reference): B=4, S=4096, D=1024, E=8
#define BB 4
#define SS 4096
#define DD 1024
#define EE 8

typedef __bf16 bf16;

#if defined(__gfx1250__)

typedef __attribute__((ext_vector_type(16))) __bf16 v16bf;
typedef __attribute__((ext_vector_type(8)))  float  v8f;
typedef __attribute__((ext_vector_type(4)))  unsigned int v4u;
typedef __attribute__((ext_vector_type(8)))  int v8i_;
typedef __attribute__((ext_vector_type(4)))  int v4i_;

#if __has_builtin(__builtin_amdgcn_tensor_load_to_lds) && __has_builtin(__builtin_amdgcn_s_wait_tensorcnt)
#define USE_TDM 1
#endif

// ---------------- math helpers (fp32, stable) ----------------
__device__ __forceinline__ float softplusf(float x) {
  // log(1+exp(x)) = max(x,0) + log1p(exp(-|x|))
  return fmaxf(x, 0.0f) + log1pf(__expf(-fabsf(x)));
}
__device__ __forceinline__ float log_gf(float x) {
  // x>=0: log(relu(x)+0.5) = log(x+0.5); x<0: -softplus(-x)
  return (x >= 0.0f) ? __logf(x + 0.5f) : -softplusf(-x);
}
__device__ __forceinline__ float logaddexpf_(float p, float q) {
  float m = fmaxf(p, q);
  return m + __logf(__expf(p - m) + __expf(q - m));
}

// ---------------- WMMA fragment load from LDS ----------------
// Tile in LDS is [16 rows][32 k] bf16 (row stride 32). 16-bit A/B layout:
// lanes 0-15 hold rows 0-15 with K slices {0..7,16..23}; lanes 16-31 hold
// the same rows with K slices {8..15,24..31}. Two contiguous 16B ds loads.
__device__ __forceinline__ v16bf load_frag16(const bf16* tile, int lane) {
  const int r   = lane & 15;
  const int klo = (lane >> 4) * 8;
  union { v16bf v; uint4 q[2]; } f;
  f.q[0] = *(const uint4*)(tile + r * 32 + klo);
  f.q[1] = *(const uint4*)(tile + r * 32 + 16 + klo);
  return f.v;
}

#if defined(USE_TDM)
// ---------------- Tensor Data Mover: 2-D tile load to LDS ----------------
// D# per CDNA5 ISA. data_size=1 (2-byte bf16).
// tile_d0 = contiguous elements (K), tile_d1 = rows; stride0 in elements.
// 6-arg builtin: (uint32x4 g0, int32x8 g1, int32x4, int32x4, int32x8, i32 cpol)
__device__ __forceinline__ void tdm_load_2d(uint32_t lds_off, const bf16* gptr,
                                            uint32_t tensor_d0, uint32_t tensor_d1,
                                            uint32_t tile_d0, uint32_t tile_d1,
                                            uint32_t stride0) {
  const uint64_t ga = (uint64_t)(uintptr_t)gptr;
  v4u g0;
  g0[0] = 1u;                                            // count=1, user mode
  g0[1] = lds_off;                                       // lds_addr (bytes)
  g0[2] = (uint32_t)ga;                                  // global_addr[31:0]
  g0[3] = (uint32_t)((ga >> 32) & 0x1FFFFFFu) | (2u << 30); // addr[56:32] | type=2
  v8i_ g1;
  g1[0] = (int)(1u << 16);                               // data_size=1 (2B)
  g1[1] = (int)((tensor_d0 & 0xFFFFu) << 16);            // tensor_dim0[15:0]
  g1[2] = (int)((tensor_d0 >> 16) | ((tensor_d1 & 0xFFFFu) << 16));
  g1[3] = (int)((tensor_d1 >> 16) | (tile_d0 << 16));    // tile_dim0
  g1[4] = (int)(tile_d1 & 0xFFFFu);                      // tile_dim1 (tile_dim2=0)
  g1[5] = (int)stride0;                                  // tensor_dim0_stride[31:0]
  g1[6] = 0;
  g1[7] = 0;
  v4i_ z4 = {0, 0, 0, 0};
  v8i_ z8 = {0, 0, 0, 0, 0, 0, 0, 0};
  __builtin_amdgcn_tensor_load_to_lds(g0, g1, z4, z4, z8, 0);
}
#endif // USE_TDM

#define WMMA_BF16(A, Bm, C) \
  __builtin_amdgcn_wmma_f32_16x16x32_bf16(false, (A), false, (Bm), (short)0, (C), false, false)

#endif // __gfx1250__

// =====================================================================
// K0: zero the router accumulators
// =====================================================================
__global__ void zero_kernel(float* __restrict__ p, int n) {
#if defined(__gfx1250__)
  int i = blockIdx.x * 256 + threadIdx.x;
  if (i < n) p[i] = 0.0f;
#endif
}

// =====================================================================
// K1: RMSNorm row (b,s): x = in * rsqrt(mean(in^2)+eps) * nw ; bf16 out
//     + atomic accumulation of column sums for the router mean.
// =====================================================================
__global__ __launch_bounds__(256) void rmsnorm_kernel(const float* __restrict__ in,
                                                      const float* __restrict__ nw,
                                                      bf16* __restrict__ x,
                                                      float* __restrict__ colsum) {
#if defined(__gfx1250__)
  const int row = blockIdx.x;          // b*S + s
  const int b = row >> 12;             // /4096
  const float* r = in + (size_t)row * DD;
  float v[4];
  float ss = 0.0f;
#pragma unroll
  for (int i = 0; i < 4; ++i) {
    v[i] = r[threadIdx.x + i * 256];
    ss += v[i] * v[i];
  }
#pragma unroll
  for (int o = 16; o > 0; o >>= 1) ss += __shfl_xor(ss, o, 32);
  __shared__ float red[8];
  if ((threadIdx.x & 31) == 0) red[threadIdx.x >> 5] = ss;
  __syncthreads();
  float tot = 0.0f;
#pragma unroll
  for (int i = 0; i < 8; ++i) tot += red[i];
  const float rs = rsqrtf(tot * (1.0f / (float)DD) + 1e-6f);
#pragma unroll
  for (int i = 0; i < 4; ++i) {
    const int d = threadIdx.x + i * 256;
    const float xv = v[i] * rs * nw[d];
    x[(size_t)row * DD + d] = (bf16)xv;
    atomicAdd(&colsum[b * DD + d], xv);
  }
#endif
}

// =====================================================================
// K2: router logits + softmax over E=8 (tiny, one block)
// =====================================================================
__global__ void router_kernel(const float* __restrict__ colsum,
                              const float* __restrict__ rw,
                              float* __restrict__ probs) {
#if defined(__gfx1250__)
  __shared__ float lg[BB * EE];
  const int t = threadIdx.x;
  if (t < BB * EE) {
    const int b = t >> 3, e = t & 7;
    float acc = 0.0f;
    for (int d = 0; d < DD; ++d)
      acc += (colsum[b * DD + d] * (1.0f / (float)SS)) * rw[d * EE + e];
    lg[t] = acc;
  }
  __syncthreads();
  if (t < BB) {
    float mx = -3.0e38f;
#pragma unroll
    for (int e = 0; e < EE; ++e) mx = fmaxf(mx, lg[t * EE + e]);
    float s = 0.0f;
#pragma unroll
    for (int e = 0; e < EE; ++e) s += __expf(lg[t * EE + e] - mx);
    const float inv = 1.0f / s;
#pragma unroll
    for (int e = 0; e < EE; ++e) probs[t * EE + e] = __expf(lg[t * EE + e] - mx) * inv;
  }
#endif
}

// =====================================================================
// K3: mix expert weights, stored TRANSPOSED (k-contiguous) in bf16:
//     outT[b][n][d] = sum_e probs[b,e] * w[e][d][n]   (w: E x Drows x N)
// =====================================================================
__global__ __launch_bounds__(256) void mix_kernel(const float* __restrict__ probs,
                                                  const float* __restrict__ w,
                                                  bf16* __restrict__ outT,
                                                  int Drows, int N) {
#if defined(__gfx1250__)
  const size_t total = (size_t)BB * N * Drows;
  const size_t step = (size_t)gridDim.x * 256;
  for (size_t i = (size_t)blockIdx.x * 256 + threadIdx.x; i < total; i += step) {
    const int b = (int)(i / ((size_t)N * Drows));
    const size_t rr = i % ((size_t)N * Drows);
    const int d = (int)(rr / N);
    const int n = (int)(rr % N);          // adjacent threads -> coalesced w reads
    float acc = 0.0f;
#pragma unroll
    for (int e = 0; e < EE; ++e)
      acc += probs[b * EE + e] * w[((size_t)e * Drows + d) * N + n];
    outT[((size_t)b * N + n) * Drows + d] = (bf16)acc;
  }
#endif
}

// =====================================================================
// K4: GEMM1 fused: hg = x @ Wi, epilogue -> lc = -softplus(gate),
//     lv = -softplus(-gate) + log_g(hidden).  Each block computes a 64x64
//     hidden tile AND its paired 64x64 gate tile (cols d and d+1024).
//     Double-buffered TDM pipeline: waves 0/1/2 each own one tile stream
//     (A, B_hidden, B_gate) and prefetch K-slice k+1 into buf^1 while all
//     waves run WMMAs on buf.  Wave tiling: 2 M-tiles x 1 N-tile x {h,g}
//     per wave -> hidden & gate accumulators for the same (s,d) live in
//     the same registers for the log-space epilogue.
// =====================================================================
__global__ __launch_bounds__(256) void gemm_hg_kernel(const bf16* __restrict__ X,
                                                      const bf16* __restrict__ WiT,
                                                      float* __restrict__ lc,
                                                      float* __restrict__ lv) {
#if defined(__gfx1250__)
  __shared__ __align__(16) bf16 sA[2][64 * 32];
  __shared__ __align__(16) bf16 sBh[2][64 * 32];
  __shared__ __align__(16) bf16 sBg[2][64 * 32];
  const int b = blockIdx.z;
  const int m0 = blockIdx.y * 64;
  const int n0 = blockIdx.x * 64;
  const bf16* Xb = X + (size_t)b * SS * DD;
  const bf16* Wb = WiT + (size_t)b * (2 * DD) * DD;
  const int tid = threadIdx.x, wave = tid >> 5, lane = tid & 31;
  const int wmA = (wave & 1) * 2;      // M-tiles {wmA, wmA+1}
  const int wn  = wave >> 1;           // N-tile 0..3
  v8f ah0 = {0,0,0,0,0,0,0,0}, ah1 = {0,0,0,0,0,0,0,0};
  v8f ag0 = {0,0,0,0,0,0,0,0}, ag1 = {0,0,0,0,0,0,0,0};

#if defined(USE_TDM)
  // prologue: fill buffer 0 with K-slice 0
  if (wave == 0)
    tdm_load_2d((uint32_t)(uintptr_t)&sA[0][0],  Xb + (size_t)m0 * DD,        DD, SS,     32u, 64u, DD);
  else if (wave == 1)
    tdm_load_2d((uint32_t)(uintptr_t)&sBh[0][0], Wb + (size_t)n0 * DD,        DD, 2 * DD, 32u, 64u, DD);
  else if (wave == 2)
    tdm_load_2d((uint32_t)(uintptr_t)&sBg[0][0], Wb + (size_t)(DD + n0) * DD, DD, 2 * DD, 32u, 64u, DD);
  int cur = 0;
  for (int k0 = 0; k0 < DD; k0 += 32) {
    if (wave < 3) __builtin_amdgcn_s_wait_tensorcnt(0);   // buf[cur] ready
    __syncthreads();
    const int kn = k0 + 32;
    if (kn < DD) {                                        // prefetch next slice
      if (wave == 0)
        tdm_load_2d((uint32_t)(uintptr_t)&sA[cur ^ 1][0],  Xb + (size_t)m0 * DD + kn,        DD, SS,     32u, 64u, DD);
      else if (wave == 1)
        tdm_load_2d((uint32_t)(uintptr_t)&sBh[cur ^ 1][0], Wb + (size_t)n0 * DD + kn,        DD, 2 * DD, 32u, 64u, DD);
      else if (wave == 2)
        tdm_load_2d((uint32_t)(uintptr_t)&sBg[cur ^ 1][0], Wb + (size_t)(DD + n0) * DD + kn, DD, 2 * DD, 32u, 64u, DD);
    }
    const v16bf a0f = load_frag16(&sA[cur][0] + (wmA + 0) * 16 * 32, lane);
    const v16bf a1f = load_frag16(&sA[cur][0] + (wmA + 1) * 16 * 32, lane);
    const v16bf bhf = load_frag16(&sBh[cur][0] + wn * 16 * 32, lane);
    const v16bf bgf = load_frag16(&sBg[cur][0] + wn * 16 * 32, lane);
    ah0 = WMMA_BF16(a0f, bhf, ah0);
    ah1 = WMMA_BF16(a1f, bhf, ah1);
    ag0 = WMMA_BF16(a0f, bgf, ag0);
    ag1 = WMMA_BF16(a1f, bgf, ag1);
    __syncthreads();                                      // done reading buf[cur]
    cur ^= 1;
  }
#else
  for (int k0 = 0; k0 < DD; k0 += 32) {
    __syncthreads();
    {
      const int row = tid >> 2, kq = (tid & 3) * 8;
      *(uint4*)(&sA[0][0]  + row * 32 + kq) = *(const uint4*)(Xb + (size_t)(m0 + row) * DD + k0 + kq);
      *(uint4*)(&sBh[0][0] + row * 32 + kq) = *(const uint4*)(Wb + (size_t)(n0 + row) * DD + k0 + kq);
      *(uint4*)(&sBg[0][0] + row * 32 + kq) = *(const uint4*)(Wb + (size_t)(DD + n0 + row) * DD + k0 + kq);
    }
    __syncthreads();
    const v16bf a0f = load_frag16(&sA[0][0] + (wmA + 0) * 16 * 32, lane);
    const v16bf a1f = load_frag16(&sA[0][0] + (wmA + 1) * 16 * 32, lane);
    const v16bf bhf = load_frag16(&sBh[0][0] + wn * 16 * 32, lane);
    const v16bf bgf = load_frag16(&sBg[0][0] + wn * 16 * 32, lane);
    ah0 = WMMA_BF16(a0f, bhf, ah0);
    ah1 = WMMA_BF16(a1f, bhf, ah1);
    ag0 = WMMA_BF16(a0f, bgf, ag0);
    ag1 = WMMA_BF16(a1f, bgf, ag1);
  }
#endif

  // epilogue: C layout VGPR r -> row r (lanes<16) / r+8 (lanes>=16), col=lane&15
  const int col = lane & 15, rhi = (lane >> 4) * 8;
  const int dd = n0 + wn * 16 + col;
  v8f AH[2] = {ah0, ah1}, AG[2] = {ag0, ag1};
#pragma unroll
  for (int mt = 0; mt < 2; ++mt) {
#pragma unroll
    for (int r = 0; r < 8; ++r) {
      const int s = m0 + (wmA + mt) * 16 + rhi + r;
      const size_t idx = ((size_t)b * SS + s) * DD + dd;
      const float hid = AH[mt][r], gt = AG[mt][r];
      lc[idx] = -softplusf(gt);
      lv[idx] = -softplusf(-gt) + log_gf(hid);
    }
  }
#endif
}

// =====================================================================
// K5a/b/c: chunked log-space scan over S (32 chunks x 128 steps).
// Recurrence: log_h_s = logaddexp(lc_s + log_h_{s-1}, lv_s).
// Chunk transform composes to x -> logaddexp(A + x, V).
// =====================================================================
#define NCH 32
#define CHL 128

__global__ __launch_bounds__(256) void scan_chunk_kernel(const float* __restrict__ lc,
                                                         const float* __restrict__ lv,
                                                         float* __restrict__ SA,
                                                         float* __restrict__ SV) {
#if defined(__gfx1250__)
  const int t = blockIdx.x * 256 + threadIdx.x;   // (b, c, d)
  if (t >= BB * NCH * DD) return;
  const int b = t / (NCH * DD);
  const int c = (t / DD) % NCH;
  const int d = t % DD;
  float A = 0.0f, V = -3.0e38f;
  const size_t base = ((size_t)b * SS + (size_t)c * CHL) * DD + d;
  for (int s = 0; s < CHL; ++s) {
    const size_t idx = base + (size_t)s * DD;
    const float a = lc[idx], v = lv[idx];
    A += a;
    V = logaddexpf_(V + a, v);
  }
  SA[t] = A;
  SV[t] = V;
#endif
}

__global__ __launch_bounds__(256) void scan_combine_kernel(const float* __restrict__ SA,
                                                           const float* __restrict__ SV,
                                                           const float* __restrict__ cur,
                                                           float* __restrict__ ch0,
                                                           float* __restrict__ new_state,
                                                           float* __restrict__ aux) {
#if defined(__gfx1250__)
  const int t = blockIdx.x * 256 + threadIdx.x;   // (b, d)
  if (t >= BB * DD) return;
  const int b = t / DD, d = t % DD;
  float lh = __logf(cur[t]);                      // current_state (B,1,D)
  for (int c = 0; c < NCH; ++c) {
    const size_t i = ((size_t)b * NCH + c) * DD + d;
    ch0[i] = lh;                                  // state entering chunk c
    lh = logaddexpf_(SA[i] + lh, SV[i]);
  }
  new_state[t] = __expf(lh);
  if (t == 0) aux[0] = 0.0f;
#endif
}

__global__ __launch_bounds__(256) void scan_apply_kernel(const float* __restrict__ lc,
                                                         const float* __restrict__ lv,
                                                         const float* __restrict__ ch0,
                                                         bf16* __restrict__ h) {
#if defined(__gfx1250__)
  const int t = blockIdx.x * 256 + threadIdx.x;   // (b, c, d)
  if (t >= BB * NCH * DD) return;
  const int b = t / (NCH * DD);
  const int c = (t / DD) % NCH;
  const int d = t % DD;
  float lh = ch0[t];
  const size_t base = ((size_t)b * SS + (size_t)c * CHL) * DD + d;
  for (int s = 0; s < CHL; ++s) {
    const size_t idx = base + (size_t)s * DD;
    lh = logaddexpf_(lc[idx] + lh, lv[idx]);
    h[idx] = (bf16)__expf(lh);
  }
#endif
}

// =====================================================================
// K6: GEMM2: out = h @ Wo + inputs.  h bf16 (S x 1024), WoT bf16
//     (1024 x 1024, k-contiguous). 64x64 tile, double-buffered TDM.
// =====================================================================
__global__ __launch_bounds__(256) void gemm_out_kernel(const bf16* __restrict__ H,
                                                       const bf16* __restrict__ WoT,
                                                       const float* __restrict__ resid,
                                                       float* __restrict__ out) {
#if defined(__gfx1250__)
  __shared__ __align__(16) bf16 sA[2][64 * 32];
  __shared__ __align__(16) bf16 sB[2][64 * 32];
  const int b = blockIdx.z;
  const int m0 = blockIdx.y * 64;
  const int n0 = blockIdx.x * 64;
  const bf16* Hb = H + (size_t)b * SS * DD;
  const bf16* Wb = WoT + (size_t)b * DD * DD;
  const int tid = threadIdx.x, wave = tid >> 5, lane = tid & 31;
  const int wmA = (wave & 1) * 2;      // M-tiles {wmA, wmA+1}
  const int wn  = wave >> 1;           // N-tile 0..3
  v8f a0 = {0,0,0,0,0,0,0,0}, a1 = {0,0,0,0,0,0,0,0};

#if defined(USE_TDM)
  if (wave == 0)
    tdm_load_2d((uint32_t)(uintptr_t)&sA[0][0], Hb + (size_t)m0 * DD, DD, SS, 32u, 64u, DD);
  else if (wave == 1)
    tdm_load_2d((uint32_t)(uintptr_t)&sB[0][0], Wb + (size_t)n0 * DD, DD, DD, 32u, 64u, DD);
  int cur = 0;
  for (int k0 = 0; k0 < DD; k0 += 32) {
    if (wave < 2) __builtin_amdgcn_s_wait_tensorcnt(0);
    __syncthreads();
    const int kn = k0 + 32;
    if (kn < DD) {
      if (wave == 0)
        tdm_load_2d((uint32_t)(uintptr_t)&sA[cur ^ 1][0], Hb + (size_t)m0 * DD + kn, DD, SS, 32u, 64u, DD);
      else if (wave == 1)
        tdm_load_2d((uint32_t)(uintptr_t)&sB[cur ^ 1][0], Wb + (size_t)n0 * DD + kn, DD, DD, 32u, 64u, DD);
    }
    const v16bf a0f = load_frag16(&sA[cur][0] + (wmA + 0) * 16 * 32, lane);
    const v16bf a1f = load_frag16(&sA[cur][0] + (wmA + 1) * 16 * 32, lane);
    const v16bf bf  = load_frag16(&sB[cur][0] + wn * 16 * 32, lane);
    a0 = WMMA_BF16(a0f, bf, a0);
    a1 = WMMA_BF16(a1f, bf, a1);
    __syncthreads();
    cur ^= 1;
  }
#else
  for (int k0 = 0; k0 < DD; k0 += 32) {
    __syncthreads();
    {
      const int row = tid >> 2, kq = (tid & 3) * 8;
      *(uint4*)(&sA[0][0] + row * 32 + kq) = *(const uint4*)(Hb + (size_t)(m0 + row) * DD + k0 + kq);
      *(uint4*)(&sB[0][0] + row * 32 + kq) = *(const uint4*)(Wb + (size_t)(n0 + row) * DD + k0 + kq);
    }
    __syncthreads();
    const v16bf a0f = load_frag16(&sA[0][0] + (wmA + 0) * 16 * 32, lane);
    const v16bf a1f = load_frag16(&sA[0][0] + (wmA + 1) * 16 * 32, lane);
    const v16bf bf  = load_frag16(&sB[0][0] + wn * 16 * 32, lane);
    a0 = WMMA_BF16(a0f, bf, a0);
    a1 = WMMA_BF16(a1f, bf, a1);
  }
#endif

  const int col = lane & 15, rhi = (lane >> 4) * 8;
  const int dd = n0 + wn * 16 + col;
  v8f AC[2] = {a0, a1};
#pragma unroll
  for (int mt = 0; mt < 2; ++mt) {
#pragma unroll
    for (int r = 0; r < 8; ++r) {
      const int s = m0 + (wmA + mt) * 16 + rhi + r;
      const size_t idx = ((size_t)b * SS + s) * DD + dd;
      out[idx] = AC[mt][r] + resid[idx];
    }
  }
#endif
}

// =====================================================================
// host launcher
// =====================================================================
extern "C" void kernel_launch(void* const* d_in, const int* in_sizes, int n_in,
                              void* d_out, int out_size, void* d_ws, size_t ws_size,
                              hipStream_t stream) {
  (void)in_sizes; (void)n_in; (void)out_size; (void)ws_size;
  const float* inputs = (const float*)d_in[0];
  // d_in[1] attention_mask: all-ones, unused by the reference math
  const float* cur    = (const float*)d_in[2];
  const float* nw     = (const float*)d_in[3];
  const float* rw     = (const float*)d_in[4];
  const float* w_in   = (const float*)d_in[5];
  const float* w_out  = (const float*)d_in[6];

  float* out       = (float*)d_out;                       // (B,S,D)
  float* new_state = out + (size_t)BB * SS * DD;          // (B,1,D)
  float* aux       = new_state + (size_t)BB * DD;         // scalar

  char* ws = (char*)d_ws;
  size_t off = 0;
  bf16* x    = (bf16*)(ws + off); off += (size_t)BB * SS * DD * 2;        // 33.5 MB
  bf16* h    = (bf16*)(ws + off); off += (size_t)BB * SS * DD * 2;        // 33.5 MB
  float* lc  = (float*)(ws + off); off += (size_t)BB * SS * DD * 4;       // 67 MB
  float* lv  = (float*)(ws + off); off += (size_t)BB * SS * DD * 4;       // 67 MB
  bf16* WiT  = (bf16*)(ws + off); off += (size_t)BB * (2 * DD) * DD * 2;  // 16.8 MB
  bf16* WoT  = (bf16*)(ws + off); off += (size_t)BB * DD * DD * 2;        // 8.4 MB
  float* colsum = (float*)(ws + off); off += (size_t)BB * DD * 4;
  float* probs  = (float*)(ws + off); off += 256;
  float* SA  = (float*)(ws + off); off += (size_t)BB * NCH * DD * 4;
  float* SV  = (float*)(ws + off); off += (size_t)BB * NCH * DD * 4;
  float* CH0 = (float*)(ws + off); off += (size_t)BB * NCH * DD * 4;

  zero_kernel<<<dim3((BB * DD + BB * EE + 255) / 256), 256, 0, stream>>>(colsum, BB * DD + BB * EE);
  rmsnorm_kernel<<<dim3(BB * SS), 256, 0, stream>>>(inputs, nw, x, colsum);
  router_kernel<<<dim3(1), 64, 0, stream>>>(colsum, rw, probs);
  mix_kernel<<<dim3(8192), 256, 0, stream>>>(probs, w_in, WiT, DD, 2 * DD);
  mix_kernel<<<dim3(4096), 256, 0, stream>>>(probs, w_out, WoT, DD, DD);
  gemm_hg_kernel<<<dim3(DD / 64, SS / 64, BB), 256, 0, stream>>>(x, WiT, lc, lv);
  scan_chunk_kernel<<<dim3((BB * NCH * DD + 255) / 256), 256, 0, stream>>>(lc, lv, SA, SV);
  scan_combine_kernel<<<dim3((BB * DD + 255) / 256), 256, 0, stream>>>(SA, SV, cur, CH0, new_state, aux);
  scan_apply_kernel<<<dim3((BB * NCH * DD + 255) / 256), 256, 0, stream>>>(lc, lv, CH0, h);
  gemm_out_kernel<<<dim3(DD / 64, SS / 64, BB), 256, 0, stream>>>(h, WoT, inputs, out);
}
